// QLSTM_27582279975491
// MI455X (gfx1250) — matrix operand order repaired and checked by reading, not compiled, not run on hardware
//
#include <hip/hip_runtime.h>
#include <math.h>

typedef __attribute__((ext_vector_type(16))) _Float16 v16h;
typedef __attribute__((ext_vector_type(8)))  float    v8f;

#define NQ    4
#define BATCH 512
#define SEQ   256
#define DIN   64
#define HDIM  128
#define BQ    16      // batch rows per workgroup (WMMA M)
#define TPB   64      // 2 wave32 waves

// fast sigmoid: v_exp_f32 + v_rcp_f32 (both TRANS ops, co-execute with WMMA)
__device__ __forceinline__ float sigmoid_f(float v) {
  return __builtin_amdgcn_rcpf(1.0f + __expf(-v));
}

// branchless fast tanh: tanh(x) = sign(x) * (1 - e) * rcp(1 + e), e = exp(-2|x|)
__device__ __forceinline__ float tanh_f(float v) {
  float e = __expf(-2.0f * fabsf(v));
  float t = (1.0f - e) * __builtin_amdgcn_rcpf(1.0f + e);
  return copysignf(t, v);
}

// ---------- quantum gate helpers on a 16-amplitude register state ----------
__device__ __forceinline__ void g1_real(float* re, float* im, int m,
                                        float u00, float u01, float u10, float u11) {
#pragma unroll
  for (int a = 0; a < 16; ++a) {
    if (a & m) continue;
    const int b = a | m;
    float r0 = re[a], i0 = im[a], r1 = re[b], i1 = im[b];
    re[a] = u00 * r0 + u01 * r1; im[a] = u00 * i0 + u01 * i1;
    re[b] = u10 * r0 + u11 * r1; im[b] = u10 * i0 + u11 * i1;
  }
}

__device__ __forceinline__ void g_rz(float* re, float* im, int m, float c, float s) {
  // diag(e^{-i t/2}, e^{+i t/2});  c=cos(t/2), s=sin(t/2)
#pragma unroll
  for (int a = 0; a < 16; ++a) {
    float r = re[a], i = im[a];
    if (a & m) { re[a] = c * r - s * i; im[a] = c * i + s * r; }
    else       { re[a] = c * r + s * i; im[a] = c * i - s * r; }
  }
}

__device__ __forceinline__ void g_rx(float* re, float* im, int m, float c, float s) {
  // [[c, -i s],[-i s, c]];  c=cos(t/2), s=sin(t/2)
#pragma unroll
  for (int a = 0; a < 16; ++a) {
    if (a & m) continue;
    const int b = a | m;
    float r0 = re[a], i0 = im[a], r1 = re[b], i1 = im[b];
    re[a] = c * r0 + s * i1; im[a] = c * i0 - s * r1;
    re[b] = c * r1 + s * i0; im[b] = c * i1 - s * r0;
  }
}

__device__ __forceinline__ void g_cnot(float* re, float* im, int mc, int mt) {
#pragma unroll
  for (int a = 0; a < 16; ++a) {
    if ((a & mc) && !(a & mt)) {
      const int b = a | mt;
      float tr = re[a]; re[a] = re[b]; re[b] = tr;
      float ti = im[a]; im[a] = im[b]; im[b] = ti;
    }
  }
}

// Full 4-qubit VQC. wc/ws = precomputed cos/sin of ansatz half-angles (12 each).
__device__ void vqc_sim(const float y[NQ], const float* __restrict__ wc,
                        const float* __restrict__ ws, float out[NQ]) {
  float re[16], im[16];
#pragma unroll
  for (int a = 0; a < 16; ++a) { re[a] = 0.0f; im[a] = 0.0f; }
  re[0] = 1.0f;
  const float is2 = 0.70710678118654752440f;
#pragma unroll
  for (int q = 0; q < NQ; ++q) {
    const int m = 1 << (3 - q);                 // qubit q <-> bit (3-q)
    g1_real(re, im, m, is2, is2, is2, -is2);    // H
    float ty = 0.5f * atanf(y[q]);
    float cy = __cosf(ty), sy = __sinf(ty);
    g1_real(re, im, m, cy, -sy, sy, cy);        // RY(atan y)
    float tz = 0.5f * atanf(y[q] * y[q]);
    g_rz(re, im, m, __cosf(tz), __sinf(tz));    // RZ(atan y^2)
  }
  // CNOT ring (0,1)(1,2)(2,3)(3,0)
#pragma unroll
  for (int q = 0; q < NQ; ++q)
    g_cnot(re, im, 1 << (3 - q), 1 << (3 - ((q + 1) & 3)));
  // ansatz: per qubit RX(w[0,q]) RY(w[1,q]) RZ(w[2,q]) with precomputed trig
#pragma unroll
  for (int q = 0; q < NQ; ++q) {
    const int m = 1 << (3 - q);
    g_rx(re, im, m, wc[0 * NQ + q], ws[0 * NQ + q]);
    float cw = wc[1 * NQ + q], sw = ws[1 * NQ + q];
    g1_real(re, im, m, cw, -sw, sw, cw);
    g_rz(re, im, m, wc[2 * NQ + q], ws[2 * NQ + q]);
  }
  // <Z_q> = sum_a sign(a,q) * |s_a|^2
#pragma unroll
  for (int q = 0; q < NQ; ++q) {
    const int m = 1 << (3 - q);
    float acc = 0.0f;
#pragma unroll
    for (int a = 0; a < 16; ++a) {
      float p = re[a] * re[a] + im[a] * im[a];
      acc += (a & m) ? -p : p;
    }
    out[q] = acc;
  }
}

__global__ __launch_bounds__(TPB) void qlstm_kernel(
    const float* __restrict__ x, const float* __restrict__ W_in,
    const float* __restrict__ b_in, const float* __restrict__ W_out,
    const float* __restrict__ b_out, const float* __restrict__ w_f,
    const float* __restrict__ w_i, const float* __restrict__ w_u,
    const float* __restrict__ w_o, float* __restrict__ out) {
  __shared__ float h_lds[BQ][HDIM];
  __shared__ float c_lds[BQ][HDIM];
  __shared__ float x_lds[BQ][DIN];
  __shared__ float y_lds[BQ][NQ];
  __shared__ float vqc_lds[4][BQ][NQ];
  __shared__ float sWin[(HDIM + DIN) * NQ];
  __shared__ float sWc[4][3 * NQ];   // cos of ansatz half-angles
  __shared__ float sWs[4][3 * NQ];   // sin of ansatz half-angles
  __shared__ float sBin[NQ];

  const int tid  = threadIdx.x;
  const int lane = tid & 31;         // wave32 lane
  const int wv   = tid >> 5;         // wave id (0..1)
  const int b0   = blockIdx.x * BQ;

  // ---- one-time setup ----
  for (int ii = tid; ii < (HDIM + DIN) * NQ; ii += TPB) sWin[ii] = W_in[ii];
  if (tid < 48) {
    int g = tid / 12, k = tid % 12;
    const float* wp = (g == 0) ? w_f : (g == 1) ? w_i : (g == 2) ? w_u : w_o;
    float half = 0.5f * wp[k];
    sWc[g][k] = __cosf(half);
    sWs[g][k] = __sinf(half);
  }
  if (tid < NQ) sBin[tid] = b_in[tid];
  for (int ii = tid; ii < BQ * HDIM; ii += TPB) {
    (&h_lds[0][0])[ii] = 0.0f;
    (&c_lds[0][0])[ii] = 0.0f;
  }

  // Constant B operands: W_out (4x128) packed split-f16 per 16-col tile.
  // B layout (16-bit 32x16): lanes 0-15 halves 0..7 = K0..7, lanes 16-31 halves 0..7 = K8..15.
  // K 0..3 = b_hi, K 4..7 = b_lo, K 8..11 = b_hi (lanes 16-31, pairs with A residual lanes).
  v16h Bop[4];
  float bo[4];
#pragma unroll
  for (int j = 0; j < 4; ++j) {
    const int nt = wv * 4 + j;
    v16h bb;
#pragma unroll
    for (int k = 0; k < 16; ++k) bb[k] = (_Float16)0.0f;
    {
      const int n = nt * 16 + (lane & 15);
#pragma unroll
      for (int k = 0; k < NQ; ++k) {
        float bv = W_out[k * HDIM + n];
        _Float16 bh = (_Float16)bv;
        _Float16 bl = (_Float16)(bv - (float)bh);
        bb[k] = bh; bb[4 + k] = bl; bb[8 + k] = bh;
      }
    }
    Bop[j] = bb;
    bo[j]  = b_out[nt * 16 + (lane & 15)];
  }
  __syncthreads();

  for (int t = 0; t < SEQ; ++t) {
    // ---- stage x_t into LDS (row-contiguous, coalesced) + prefetch next ----
    for (int ii = tid; ii < BQ * DIN; ii += TPB) {
      const int r = ii >> 6, d = ii & 63;
      x_lds[r][d] = x[(size_t)(b0 + r) * SEQ * DIN + (size_t)t * DIN + d];
    }
    if (t + 1 < SEQ && tid < BQ)
      __builtin_prefetch(&x[(size_t)(b0 + tid) * SEQ * DIN + (size_t)(t + 1) * DIN], 0, 1);
    __syncthreads();

    // ---- y = [h | x_t] @ W_in + b_in : one (row, q) per thread ----
    {
      const int r = tid >> 2, q = tid & 3;
      float acc = sBin[q];
#pragma unroll 8
      for (int k = 0; k < HDIM; ++k) acc += h_lds[r][k] * sWin[k * NQ + q];
#pragma unroll 8
      for (int d = 0; d < DIN; ++d) acc += x_lds[r][d] * sWin[(HDIM + d) * NQ + q];
      y_lds[r][q] = acc;
    }
    __syncthreads();

    // ---- 4 VQCs x 16 rows: one full simulation per thread ----
    {
      const int g = tid >> 4, r = tid & 15;
      float yv[NQ], o4[NQ];
#pragma unroll
      for (int q = 0; q < NQ; ++q) yv[q] = y_lds[r][q];
      vqc_sim(yv, sWc[g], sWs[g], o4);
#pragma unroll
      for (int q = 0; q < NQ; ++q) vqc_lds[g][r][q] = o4[q];
    }
    __syncthreads();

    // ---- A operands: vqc (16x4) split-f16, built branch-free.
    // A layout (16-bit 16x32): lanes 0-15 halves 0..7 = K0..7 (hi twice),
    // lanes 16-31 halves 0..3 = K8..11 (residual lo).
    const bool loHalf = (lane < 16);
    const int  ar     = lane & 15;
    v16h Ag[4];
#pragma unroll
    for (int g = 0; g < 4; ++g) {
      v16h a;
#pragma unroll
      for (int k = 0; k < 16; ++k) a[k] = (_Float16)0.0f;
#pragma unroll
      for (int k = 0; k < NQ; ++k) {
        float fv = vqc_lds[g][ar][k];
        _Float16 hi = (_Float16)fv;
        _Float16 lo = (_Float16)(fv - (float)hi);
        a[k]     = loHalf ? hi : lo;                 // K0..3 | K8..11
        a[4 + k] = loHalf ? hi : (_Float16)0.0f;     // K4..7 | K12..15
      }
      Ag[g] = a;
    }

    // ---- WMMA gate GEMMs + LSTM cell update (this wave owns 4 N-tiles) ----
#pragma unroll
    for (int j = 0; j < 4; ++j) {
      const int nt = wv * 4 + j;
      v8f aF, aI, aG, aO;
#pragma unroll
      for (int v = 0; v < 8; ++v) { aF[v] = bo[j]; aI[v] = bo[j]; aG[v] = bo[j]; aO[v] = bo[j]; }
      aF = __builtin_amdgcn_wmma_f32_16x16x32_f16(false, Ag[0], false, Bop[j], (short)0, aF, false, false);
      aI = __builtin_amdgcn_wmma_f32_16x16x32_f16(false, Ag[1], false, Bop[j], (short)0, aI, false, false);
      aG = __builtin_amdgcn_wmma_f32_16x16x32_f16(false, Ag[2], false, Bop[j], (short)0, aG, false, false);
      aO = __builtin_amdgcn_wmma_f32_16x16x32_f16(false, Ag[3], false, Bop[j], (short)0, aO, false, false);
      const int hi8 = (lane >> 4) << 3;          // D layout: lanes 16-31 -> rows 8..15
      const int n   = nt * 16 + (lane & 15);
#pragma unroll
      for (int v = 0; v < 8; ++v) {
        const int r = v + hi8;
        float fg = sigmoid_f(aF[v]);
        float ig = sigmoid_f(aI[v]);
        float gg = tanh_f(aG[v]);
        float og = sigmoid_f(aO[v]);
        float cn = fg * c_lds[r][n] + ig * gg;
        float hn = og * tanh_f(cn);
        c_lds[r][n] = cn;
        h_lds[r][n] = hn;
        out[(size_t)(b0 + r) * SEQ * HDIM + (size_t)t * HDIM + n] = hn;
      }
    }
    __syncthreads();
  }

  // ---- hT, cT (concatenated after outs) ----
  const size_t outs = (size_t)BATCH * SEQ * HDIM;
  for (int ii = tid; ii < BQ * HDIM; ii += TPB) {
    const int r = ii >> 7, n = ii & 127;
    out[outs + (size_t)(b0 + r) * HDIM + n] = h_lds[r][n];
    out[outs + (size_t)BATCH * HDIM + (size_t)(b0 + r) * HDIM + n] = c_lds[r][n];
  }
}

extern "C" void kernel_launch(void* const* d_in, const int* in_sizes, int n_in,
                              void* d_out, int out_size, void* d_ws, size_t ws_size,
                              hipStream_t stream) {
  (void)in_sizes; (void)n_in; (void)out_size; (void)d_ws; (void)ws_size;
  const float* x     = (const float*)d_in[0];
  const float* W_in  = (const float*)d_in[1];
  const float* b_in  = (const float*)d_in[2];
  const float* W_out = (const float*)d_in[3];
  const float* b_out = (const float*)d_in[4];
  const float* w_f   = (const float*)d_in[5];
  const float* w_i   = (const float*)d_in[6];
  const float* w_u   = (const float*)d_in[7];
  const float* w_o   = (const float*)d_in[8];
  float* out = (float*)d_out;
  qlstm_kernel<<<dim3(BATCH / BQ), dim3(TPB), 0, stream>>>(
      x, W_in, b_in, W_out, b_out, w_f, w_i, w_u, w_o, out);
}
